// MultiHeadAttention_44169443672179
// MI455X (gfx1250) — matrix operand order repaired and checked by reading, not compiled, not run on hardware
//
#include <hip/hip_runtime.h>
#include <hip/hip_bf16.h>

typedef __attribute__((ext_vector_type(16))) _Float16 v16h;
typedef __attribute__((ext_vector_type(8)))  _Float16 v8h;
typedef __attribute__((ext_vector_type(8)))  float    v8f;

union V16 { v16h v; v8h h[2]; };

#define WMMA_F16(A, B, C) \
  __builtin_amdgcn_wmma_f32_16x16x32_f16(false, (A), false, (B), (short)0, (C), false, false)

static constexpr int Bsz = 2;
static constexpr int S   = 2048;
static constexpr int D   = 1024;
static constexpr int H   = 16;
static constexpr int DH  = 64;
static constexpr int M   = Bsz * S;          // 4096 rows
static constexpr int WSZ = D * D;            // one weight matrix, elements
static constexpr int XSZ = M * D;            // x / q / k / v / ctx element count

// ---------------------------------------------------------------------------
// CDNA5 async copy engine: global -> LDS without the VGPR round-trip.
// LDS destination = low 32 bits of the generic pointer.
// Tracked by ASYNCcnt; completion enforced with s_wait_asynccnt.
// ---------------------------------------------------------------------------
__device__ inline void async_ld_b128(const _Float16* g, const _Float16* l) {
  unsigned long long ga = (unsigned long long)(uintptr_t)g;
  unsigned int       la = (unsigned int)(uintptr_t)l;
  asm volatile("global_load_async_to_lds_b128 %0, %1, off"
               :: "v"(la), "v"(ga) : "memory");
}
__device__ inline void wait_async0() {
  asm volatile("s_wait_asynccnt 0x0" ::: "memory");
}

__device__ inline float rowmax16(float v) {
  v = fmaxf(v, __shfl_xor(v, 1));
  v = fmaxf(v, __shfl_xor(v, 2));
  v = fmaxf(v, __shfl_xor(v, 4));
  v = fmaxf(v, __shfl_xor(v, 8));
  return v;
}
__device__ inline float rowsum16(float v) {
  v += __shfl_xor(v, 1);
  v += __shfl_xor(v, 2);
  v += __shfl_xor(v, 4);
  v += __shfl_xor(v, 8);
  return v;
}

// ---------------------------------------------------------------------------
// fp32 -> f16 conversion; weights transposed to [N][K] so GEMM B operand
// loads contiguous rows and matches the WMMA B-fragment K-striping.
// ---------------------------------------------------------------------------
__global__ __launch_bounds__(256) void convert_kernel(
    const float* __restrict__ x,
    const float* __restrict__ Wq, const float* __restrict__ Wk,
    const float* __restrict__ Wv, const float* __restrict__ Wo,
    _Float16* __restrict__ xh, _Float16* __restrict__ wt) {
  int i = blockIdx.x * blockDim.x + threadIdx.x;
  if (i < XSZ) xh[i] = (_Float16)x[i];
  if (i < WSZ) {
    int k = i & (D - 1);
    int n = i >> 10;
    wt[0 * WSZ + n * D + k] = (_Float16)Wq[k * D + n];
    wt[1 * WSZ + n * D + k] = (_Float16)Wk[k * D + n];
    wt[2 * WSZ + n * D + k] = (_Float16)Wv[k * D + n];
    wt[3 * WSZ + n * D + k] = (_Float16)Wo[k * D + n];
  }
}

// ---------------------------------------------------------------------------
// Tiled WMMA GEMM: C[M=4096, N=1024] = A[M,K=1024] * Bt[N,K]^T
// Block tile 128x128, BK=64, 8 waves (4x2), wave tile 32x64.
// 16 WMMAs per synchronization point; double-buffered LDS (72 KB total)
// filled by the async copy engine (ASYNCcnt pipeline):
//   wait(cur) -> barrier -> issue(next, alt buffer) -> 16x WMMA from cur.
// kind==0 : QKV projection, blockIdx.z selects weight; stores f16 into
//           Q[b,h,s,d] / K[b,h,s,d] / Vt[b,h,d,s].
// kind==1 : output projection, stores fp32 + bias into d_out.
// ---------------------------------------------------------------------------
__global__ __launch_bounds__(256) void gemm_kernel(
    const _Float16* __restrict__ A, const _Float16* __restrict__ Bt,
    _Float16* __restrict__ oq, _Float16* __restrict__ ok,
    _Float16* __restrict__ ovt, float* __restrict__ of,
    const float* __restrict__ bias, int kind) {
  constexpr int LST = 72;  // LDS row stride in halves (144B, 16B-aligned rows)
  __shared__ __align__(16) _Float16 As[2][128 * LST];
  __shared__ __align__(16) _Float16 Bs[2][128 * LST];

  const int n0 = blockIdx.x * 128;
  const int m0 = blockIdx.y * 128;
  const int wz = blockIdx.z;
  if (kind == 0) Bt += (size_t)wz * WSZ;

  const int tid  = threadIdx.x;
  const int wave = tid >> 5;
  const int lane = tid & 31;
  const int hb   = lane >> 4;   // half of wave (k-striping select)
  const int lr   = lane & 15;
  const int waveM = wave >> 1;  // 0..3
  const int waveN = wave & 1;   // 0..1

  v8f acc[2][4] = {};

  // cooperative-load geometry: 128 rows x 64 halves per matrix per stage,
  // 256 threads -> 4 b128 async issues per thread per matrix
  const int ldr = tid >> 1;          // 0..127
  const int ldc = (tid & 1) * 32;    // 0 or 32 halves
  const _Float16* agp = A  + (m0 + ldr) * D + ldc;   // + k0 per stage
  const _Float16* bgp = Bt + (n0 + ldr) * D + ldc;
  const int lofs = ldr * LST + ldc;

  // prologue: async-fill k-slab 0 into buffer 0
#pragma unroll
  for (int c = 0; c < 32; c += 8) {
    async_ld_b128(agp + c, &As[0][lofs + c]);
    async_ld_b128(bgp + c, &Bs[0][lofs + c]);
  }

  for (int k0 = 0; k0 < D; k0 += 64) {
    const int cur = (k0 >> 6) & 1;
    wait_async0();       // this thread's async fills for buffer `cur` are done
    __syncthreads();     // everyone's are done; prior reads of `cur^1` retired

    if (k0 + 64 < D) {   // overlap next slab fill with WMMA on current slab
      const int nxt = cur ^ 1;
#pragma unroll
      for (int c = 0; c < 32; c += 8) {
        async_ld_b128(agp + k0 + 64 + c, &As[nxt][lofs + c]);
        async_ld_b128(bgp + k0 + 64 + c, &Bs[nxt][lofs + c]);
      }
      if (k0 + 128 < D) {  // L2 prefetch two slabs ahead
        __builtin_prefetch(agp + k0 + 128, 0, 2);
        __builtin_prefetch(bgp + k0 + 128, 0, 2);
      }
    }

    V16 af[2][2], bf[4][2];
#pragma unroll
    for (int mt = 0; mt < 2; ++mt) {
      const _Float16* ar = &As[cur][(waveM * 32 + mt * 16 + lr) * LST];
      af[mt][0].h[0] = *(const v8h*)(ar + 8 * hb);        // k = 8h..
      af[mt][0].h[1] = *(const v8h*)(ar + 16 + 8 * hb);   // k = 16+8h..
      af[mt][1].h[0] = *(const v8h*)(ar + 32 + 8 * hb);   // k = 32+8h..
      af[mt][1].h[1] = *(const v8h*)(ar + 48 + 8 * hb);   // k = 48+8h..
    }
#pragma unroll
    for (int nt = 0; nt < 4; ++nt) {
      const _Float16* br = &Bs[cur][(waveN * 64 + nt * 16 + lr) * LST];
      bf[nt][0].h[0] = *(const v8h*)(br + 8 * hb);        // k = 8h..
      bf[nt][0].h[1] = *(const v8h*)(br + 16 + 8 * hb);
      bf[nt][1].h[0] = *(const v8h*)(br + 32 + 8 * hb);   // k = 32+8h..
      bf[nt][1].h[1] = *(const v8h*)(br + 48 + 8 * hb);
    }
#pragma unroll
    for (int mt = 0; mt < 2; ++mt)
#pragma unroll
      for (int nt = 0; nt < 4; ++nt) {
        acc[mt][nt] = WMMA_F16(af[mt][0].v, bf[nt][0].v, acc[mt][nt]);
        acc[mt][nt] = WMMA_F16(af[mt][1].v, bf[nt][1].v, acc[mt][nt]);
      }
  }

  const int mbase = m0 + waveM * 32;
  const int nbase = n0 + waveN * 64;
#pragma unroll
  for (int mt = 0; mt < 2; ++mt)
#pragma unroll
    for (int nt = 0; nt < 4; ++nt)
#pragma unroll
      for (int r = 0; r < 8; ++r) {
        const int row = mbase + mt * 16 + r + 8 * hb;   // C layout: m = r + 8*half
        const int col = nbase + nt * 16 + lr;           // n = lane&15
        const float val = acc[mt][nt][r];
        if (kind == 1) {
          of[row * D + col] = val + bias[col];
        } else {
          const int b = row >> 11, s = row & (S - 1);
          const int h = col >> 6,  d = col & (DH - 1);
          if (wz == 0)      oq[((b * H + h) * S + s) * DH + d]  = (_Float16)val;
          else if (wz == 1) ok[((b * H + h) * S + s) * DH + d]  = (_Float16)val;
          else              ovt[((b * H + h) * DH + d) * S + s] = (_Float16)val;
        }
      }
}

// ---------------------------------------------------------------------------
// Flash attention: one block = one (b,h) x 128 query rows. 8 waves, each wave
// owns 16 q rows. kv tiles of 64 staged into double-buffered LDS by the async
// copy engine (16 WMMAs per sync: 8 scores + 8 P*V); online softmax with
// 16-lane xor-shuffle row reductions; P is re-laid out through a per-wave LDS
// tile to become a WMMA A fragment for P*V.
// ---------------------------------------------------------------------------
__global__ __launch_bounds__(256) void attn_kernel(
    const _Float16* __restrict__ Qh, const _Float16* __restrict__ Kh,
    const _Float16* __restrict__ Vth, _Float16* __restrict__ ctxh) {
  constexpr int KST = 72;  // K tile row stride in halves (144B)
  constexpr int VST = 72;  // V tile row stride
  constexpr int PST = 72;  // P tile row stride
  __shared__ __align__(16) _Float16 Ks[2][64 * KST];     // [kv 0..63][d 0..63]
  __shared__ __align__(16) _Float16 Vs[2][64 * VST];     // [d 0..63][kv 0..63]
  __shared__ __align__(16) _Float16 Ps[8 * 16 * PST];    // per-wave 16x64

  const int bh = blockIdx.y;      // b*H + h
  const int qb = blockIdx.x;      // query block (128 rows)
  const int tid  = threadIdx.x;
  const int wave = tid >> 5;
  const int lane = tid & 31;
  const int hb   = lane >> 4;
  const int lr   = lane & 15;
  const int q0   = qb * 128 + wave * 16;

  // Q fragments (16 rows x 64 d) held in registers for the whole kv loop.
  const _Float16* qbase = Qh + ((size_t)bh * S + q0 + lr) * DH;
  V16 qa0, qa1;
  qa0.h[0] = *(const v8h*)(qbase + 8 * hb);
  qa0.h[1] = *(const v8h*)(qbase + 16 + 8 * hb);
  qa1.h[0] = *(const v8h*)(qbase + 32 + 8 * hb);
  qa1.h[1] = *(const v8h*)(qbase + 48 + 8 * hb);

  float mrow[8], lrow[8];
  v8f o[4] = {};
#pragma unroll
  for (int r = 0; r < 8; ++r) { mrow[r] = -1e30f; lrow[r] = 0.0f; }

  _Float16* pw = &Ps[wave * 16 * PST];
  const int nsteps = (qb + 1) * 2;  // causal: kv tiles of 64 up to q-block end

  // cooperative-load geometry: each tile 64 x 64 halves, 2 b128 per thread
  const int kr = tid >> 2, kc = (tid & 3) * 16;
  const _Float16* kgp = Kh  + ((size_t)bh * S  + kr) * DH + kc;  // + kv*DH
  const _Float16* vgp = Vth + ((size_t)bh * DH + kr) * S  + kc;  // + kv
  const int kofs = kr * KST + kc;

  // prologue: async-fill kv tile 0 into buffer 0
  async_ld_b128(kgp,     &Ks[0][kofs]);
  async_ld_b128(kgp + 8, &Ks[0][kofs + 8]);
  async_ld_b128(vgp,     &Vs[0][kofs]);
  async_ld_b128(vgp + 8, &Vs[0][kofs + 8]);

  for (int step = 0; step < nsteps; ++step) {
    const int cur = step & 1;
    wait_async0();
    __syncthreads();

    if (step + 1 < nsteps) {  // overlap next kv tile fill with this step's math
      const size_t kvn = (size_t)(step + 1) * 64;
      async_ld_b128(kgp + kvn * DH,     &Ks[cur ^ 1][kofs]);
      async_ld_b128(kgp + kvn * DH + 8, &Ks[cur ^ 1][kofs + 8]);
      async_ld_b128(vgp + kvn,          &Vs[cur ^ 1][kofs]);
      async_ld_b128(vgp + kvn + 8,      &Vs[cur ^ 1][kofs + 8]);
    }

    const int kv = step * 64;

    // scores: four 16x16 tiles (kv columns 16j .. 16j+15)
    v8f sc[4];
#pragma unroll
    for (int j = 0; j < 4; ++j) {
      const _Float16* kb = &Ks[cur][(j * 16 + lr) * KST];
      V16 kb0, kb1;
      kb0.h[0] = *(const v8h*)(kb + 8 * hb);            // d = 8h ..
      kb0.h[1] = *(const v8h*)(kb + 16 + 8 * hb);
      kb1.h[0] = *(const v8h*)(kb + 32 + 8 * hb);       // d = 32+8h ..
      kb1.h[1] = *(const v8h*)(kb + 48 + 8 * hb);
      v8f s = {};
      s = WMMA_F16(qa0.v, kb0.v, s);
      s = WMMA_F16(qa1.v, kb1.v, s);
      sc[j] = s;
    }

    // online softmax over 64 kv columns, write P (f16) to per-wave LDS tile
#pragma unroll
    for (int r = 0; r < 8; ++r) {
      const int mg = q0 + r + 8 * hb;
      float sv[4];
#pragma unroll
      for (int j = 0; j < 4; ++j) {
        sv[j] = sc[j][r] * 0.125f;
        if (kv + j * 16 + lr > mg) sv[j] = -1e30f;
      }
      float t = fmaxf(fmaxf(sv[0], sv[1]), fmaxf(sv[2], sv[3]));
      t = rowmax16(t);
      const float mnew  = fmaxf(mrow[r], t);
      const float alpha = __expf(mrow[r] - mnew);
      float psum = 0.0f;
      const int mloc = r + 8 * hb;
#pragma unroll
      for (int j = 0; j < 4; ++j) {
        const float p = __expf(sv[j] - mnew);
        psum += p;
        pw[mloc * PST + j * 16 + lr] = (_Float16)p;
      }
      lrow[r] = lrow[r] * alpha + rowsum16(psum);
      mrow[r] = mnew;
#pragma unroll
      for (int dt = 0; dt < 4; ++dt) o[dt][r] *= alpha;
    }

    // P (16x64) * V (64x64): A fragments from own wave's LDS tile
    V16 pa0, pa1;
    pa0.h[0] = *(const v8h*)(pw + lr * PST + 8 * hb);
    pa0.h[1] = *(const v8h*)(pw + lr * PST + 16 + 8 * hb);
    pa1.h[0] = *(const v8h*)(pw + lr * PST + 32 + 8 * hb);
    pa1.h[1] = *(const v8h*)(pw + lr * PST + 48 + 8 * hb);
#pragma unroll
    for (int dt = 0; dt < 4; ++dt) {
      const _Float16* vrow = &Vs[cur][(dt * 16 + lr) * VST];
      V16 vb0, vb1;
      vb0.h[0] = *(const v8h*)(vrow + 8 * hb);            // kv = 8h ..
      vb0.h[1] = *(const v8h*)(vrow + 16 + 8 * hb);
      vb1.h[0] = *(const v8h*)(vrow + 32 + 8 * hb);       // kv = 32+8h ..
      vb1.h[1] = *(const v8h*)(vrow + 48 + 8 * hb);
      o[dt] = WMMA_F16(pa0.v, vb0.v, o[dt]);
      o[dt] = WMMA_F16(pa1.v, vb1.v, o[dt]);
    }
    __syncthreads();  // keep reads of `cur` strictly before next overwrite
  }

  // normalize and write ctx as f16 [b][s][h*64+d]
  const int b = bh >> 4, h = bh & (H - 1);
#pragma unroll
  for (int dt = 0; dt < 4; ++dt)
#pragma unroll
    for (int r = 0; r < 8; ++r) {
      const int s = q0 + r + 8 * hb;
      const int d = dt * 16 + lr;
      ctxh[((size_t)b * S + s) * D + h * DH + d] = (_Float16)(o[dt][r] / lrow[r]);
    }
}

// ---------------------------------------------------------------------------
extern "C" void kernel_launch(void* const* d_in, const int* in_sizes, int n_in,
                              void* d_out, int out_size, void* d_ws, size_t ws_size,
                              hipStream_t stream) {
  const float* x  = (const float*)d_in[0];
  const float* Wq = (const float*)d_in[1];
  const float* Wk = (const float*)d_in[2];
  const float* Wv = (const float*)d_in[3];
  const float* Wo = (const float*)d_in[4];
  const float* bo = (const float*)d_in[5];
  float* out = (float*)d_out;

  _Float16* ws16 = (_Float16*)d_ws;
  _Float16* xh   = ws16;                    // 4M halves (reused as ctx later)
  _Float16* wt   = xh + XSZ;                // 4 transposed weights, 4M halves
  _Float16* qh   = wt + 4 * WSZ;            // 4M halves
  _Float16* kh   = qh + XSZ;                // 4M halves
  _Float16* vth  = kh + XSZ;                // 4M halves
  _Float16* ctxh = xh;                      // reuse: x no longer needed

  convert_kernel<<<XSZ / 256, 256, 0, stream>>>(x, Wq, Wk, Wv, Wo, xh, wt);

  // QKV projections: one launch, blockIdx.z picks the weight / output layout
  gemm_kernel<<<dim3(D / 128, M / 128, 3), 256, 0, stream>>>(
      xh, wt, qh, kh, vth, nullptr, nullptr, 0);

  attn_kernel<<<dim3(S / 128, Bsz * H), 256, 0, stream>>>(qh, kh, vth, ctxh);

  // output projection + bias -> fp32 d_out
  gemm_kernel<<<dim3(D / 128, M / 128, 1), 256, 0, stream>>>(
      ctxh, wt + 3 * WSZ, nullptr, nullptr, nullptr, out, bo, 1);
}
